// Retriever_45294725103970
// MI455X (gfx1250) — matrix operand order repaired and checked by reading, not compiled, not run on hardware
//
#include <hip/hip_runtime.h>
#include <hip/hip_bf16.h>
#include <math.h>

// ---------------------------------------------------------------------------
// CDNA5 (gfx1250) wave32 WMMA types
// ---------------------------------------------------------------------------
typedef __bf16          v16bf  __attribute__((ext_vector_type(16)));
typedef float           v8f    __attribute__((ext_vector_type(8)));
typedef unsigned short  u16x16 __attribute__((ext_vector_type(16)));
typedef unsigned short  u16x8  __attribute__((ext_vector_type(8)));

__device__ __forceinline__ v8f wmma_bf16(v16bf a, v16bf b, v8f c) {
    // D = A(16x32 bf16) * B(32x16 bf16) + C(16x16 f32)
    return __builtin_amdgcn_wmma_f32_16x16x32_bf16(
        /*neg_a=*/false, a, /*neg_b=*/false, b,
        /*c_mod=*/(short)0, c, /*reuse_a=*/false, /*reuse_b=*/false);
}

// hardware f32 -> bf16 convert (lets the backend use v_cvt_*bf16* ops)
__device__ __forceinline__ __bf16 f2bf(float f) { return (__bf16)f; }
__device__ __forceinline__ unsigned short f2bfbits(float f) {
    __bf16 h = (__bf16)f;
    return __builtin_bit_cast(unsigned short, h);
}

// B-fragment swizzle for a KxN weight (K mult of 32, N mult of 16), per ISA
// 16-bit B layout: lane L (g=L>>4, n=L&15) element e holds B[c*32+(e>>3)*16+g*8+(e&7)][t*16+n].
__device__ __forceinline__ int swz_idx(int k, int j) {
    int c = k >> 5, g = (k >> 3) & 1, e = (k & 7) | (((k >> 4) & 1) << 3);
    int t = j >> 4, n = j & 15;
    int L = g * 16 + n;
    return (((t * 2 + c) * 32 + L) << 4) + e;
}

// Load A-fragment (16x32 bf16) chunk c from a global f32 row (64 floats).
// ISA A layout: lane L (M=L&15, g=L>>4) element e -> K = c*32 + (e>>3)*16 + g*8 + (e&7)
__device__ __forceinline__ v16bf load_a_global(const float* __restrict__ rowp, int c, int g) {
    v16bf v;
#pragma unroll
    for (int blk = 0; blk < 2; ++blk) {
        const float4* p = (const float4*)(rowp + c * 32 + blk * 16 + g * 8);
        float4 f0 = p[0];
        float4 f1 = p[1];
        int eb = blk * 8;
        v[eb + 0] = f2bf(f0.x); v[eb + 1] = f2bf(f0.y);
        v[eb + 2] = f2bf(f0.z); v[eb + 3] = f2bf(f0.w);
        v[eb + 4] = f2bf(f1.x); v[eb + 5] = f2bf(f1.y);
        v[eb + 6] = f2bf(f1.z); v[eb + 7] = f2bf(f1.w);
    }
    return v;
}

// Load pre-swizzled B-fragment (contiguous 32B per lane).
__device__ __forceinline__ v16bf load_b_swz(const unsigned short* __restrict__ wsz,
                                            int t, int c, int L) {
    const u16x16* p = (const u16x16*)(wsz + ((((t * 2 + c) * 32) + L) << 4));
    return __builtin_bit_cast(v16bf, *p);
}

#define H1_STRIDE 72   // halves per row (16B-aligned segments, staggered banks)
#define ACC_STRIDE 68  // floats per row
#define ETILES 8       // 16-edge tiles per wave (128 edges/block)
#define NTILES 8       // 16-node tiles per wave

// A-fragment from bf16 LDS tile h1[16][H1_STRIDE] (row-major features).
__device__ __forceinline__ v16bf load_a_lds(const unsigned short* h1, int row, int c, int g) {
    u16x8 lo = *(const u16x8*)(h1 + row * H1_STRIDE + c * 32 + 0  + g * 8);
    u16x8 hi = *(const u16x8*)(h1 + row * H1_STRIDE + c * 32 + 16 + g * 8);
    u16x16 u;
#pragma unroll
    for (int i = 0; i < 8; ++i) { u[i] = lo[i]; u[8 + i] = hi[i]; }
    return __builtin_bit_cast(v16bf, u);
}

__device__ __forceinline__ void lds_fence() {
    asm volatile("s_wait_dscnt 0" ::: "memory");
}

// ---------------------------------------------------------------------------
// Kernel 1: fold weights. qc = (Q@Wq+bq)@W1q + b1 + br@W1r ; Wr1 = Wr@W1r ;
// Weh = We@W1h ; Wet = We@W1t ; bvh = be@W1h ; bvt = be@W1t ; W2 reswizzled.
// 1 block x 64 threads (thread = output column j).
// ---------------------------------------------------------------------------
__global__ void __launch_bounds__(64)
prep_kernel(const float* __restrict__ Q,  const float* __restrict__ Wq, const float* __restrict__ bq,
            const float* __restrict__ We, const float* __restrict__ be,
            const float* __restrict__ Wr, const float* __restrict__ br,
            const float* __restrict__ W1, const float* __restrict__ b1,
            const float* __restrict__ W2,
            float* __restrict__ qc, float* __restrict__ bvh, float* __restrict__ bvt,
            unsigned short* __restrict__ Wr1s, unsigned short* __restrict__ Wehs,
            unsigned short* __restrict__ Wets, unsigned short* __restrict__ W2s) {
    __shared__ float qre[64 * 65];
    const int j = threadIdx.x;  // 0..63

    // q_repr = Q @ Wq + bq
    for (int i = 0; i < 64; ++i) {
        float a = bq[j];
        for (int k = 0; k < 64; ++k) a += Q[i * 64 + k] * Wq[k * 64 + j];
        qre[i * 65 + j] = a;
    }
    __syncthreads();

    // constant column: b1 + br @ W1r
    float c1 = b1[j];
    for (int k = 0; k < 64; ++k) c1 += br[k] * W1[(128 + k) * 64 + j];

    // qc = q_repr @ W1q + c1
    for (int i = 0; i < 64; ++i) {
        float a = c1;
        for (int k = 0; k < 64; ++k) a += qre[i * 65 + k] * W1[k * 64 + j];
        qc[i * 64 + j] = a;
    }

    // bias vectors from be
    float ah = 0.f, at = 0.f;
    for (int k = 0; k < 64; ++k) {
        ah += be[k] * W1[(64 + k) * 64 + j];
        at += be[k] * W1[(192 + k) * 64 + j];
    }
    bvh[j] = ah; bvt[j] = at;

    // folded 64x64 weights, stored bf16 in WMMA B layout
    for (int k = 0; k < 64; ++k) {
        float r1 = 0.f, eh = 0.f, et = 0.f;
        for (int m = 0; m < 64; ++m) {
            float wr = Wr[k * 64 + m], we = We[k * 64 + m];
            r1 += wr * W1[(128 + m) * 64 + j];
            eh += we * W1[(64 + m) * 64 + j];
            et += we * W1[(192 + m) * 64 + j];
        }
        int idx = swz_idx(k, j);
        Wr1s[idx] = f2bfbits(r1);
        Wehs[idx] = f2bfbits(eh);
        Wets[idx] = f2bfbits(et);
    }

    // W2 (64x128) -> bf16 swizzled
    for (int half = 0; half < 2; ++half) {
        int jj = j + half * 64;
        for (int k = 0; k < 64; ++k)
            W2s[swz_idx(k, jj)] = f2bfbits(W2[k * 128 + jj]);
    }
}

// ---------------------------------------------------------------------------
// Kernel 2: nh = node_emb @ Weh + bvh ; nt = node_emb @ Wet + bvt
// 1 wave per NTILES 16-node tiles; weights hoisted; coalesced stores via LDS.
// ---------------------------------------------------------------------------
__global__ void __launch_bounds__(32)
node_kernel(const float* __restrict__ node_emb,
            const unsigned short* __restrict__ Wehs, const unsigned short* __restrict__ Wets,
            const float* __restrict__ bvh, const float* __restrict__ bvt,
            float* __restrict__ nh, float* __restrict__ nt, int nNodes) {
    __shared__ float nhT[16 * ACC_STRIDE];
    __shared__ float ntT[16 * ACC_STRIDE];

    const int L = threadIdx.x, g = L >> 4, n = L & 15;
    const int row2 = L >> 1, cb = (L & 1) * 32;

    // hoisted weight fragments + biases (loop-invariant)
    v16bf wehf[4][2], wetf[4][2];
    float bvhv[4], bvtv[4];
#pragma unroll
    for (int t = 0; t < 4; ++t) {
        wehf[t][0] = load_b_swz(Wehs, t, 0, L);
        wehf[t][1] = load_b_swz(Wehs, t, 1, L);
        wetf[t][0] = load_b_swz(Wets, t, 0, L);
        wetf[t][1] = load_b_swz(Wets, t, 1, L);
        bvhv[t] = bvh[t * 16 + n];
        bvtv[t] = bvt[t * 16 + n];
    }

    for (int tt = 0; tt < NTILES; ++tt) {
        const int base = (blockIdx.x * NTILES + tt) * 16;
        if (base >= nNodes) break;  // uniform

        int ar = base + n; if (ar >= nNodes) ar = nNodes - 1;  // clamp, EXEC stays full
        const float* rowp = node_emb + ar * 64;
        v16bf a0 = load_a_global(rowp, 0, g);
        v16bf a1 = load_a_global(rowp, 1, g);

#pragma unroll
        for (int t = 0; t < 4; ++t) {
            v8f dh = {}; v8f dt = {};
            dh = wmma_bf16(a0, wehf[t][0], dh);
            dh = wmma_bf16(a1, wehf[t][1], dh);
            dt = wmma_bf16(a0, wetf[t][0], dt);
            dt = wmma_bf16(a1, wetf[t][1], dt);
#pragma unroll
            for (int r = 0; r < 8; ++r) {
                int m = g * 8 + r;
                nhT[m * ACC_STRIDE + t * 16 + n] = dh[r] + bvhv[t];
                ntT[m * ACC_STRIDE + t * 16 + n] = dt[r] + bvtv[t];
            }
        }

        lds_fence();  // tile complete in LDS

        int node = base + row2;
        if (node < nNodes) {
            float* __restrict__ dh = nh + node * 64 + cb;
            float* __restrict__ dt = nt + node * 64 + cb;
#pragma unroll
            for (int jj = 0; jj < 8; ++jj) {
                *(float4*)(dh + jj * 4) = *(const float4*)&nhT[row2 * ACC_STRIDE + cb + jj * 4];
                *(float4*)(dt + jj * 4) = *(const float4*)&ntT[row2 * ACC_STRIDE + cb + jj * 4];
            }
        }
        // next iteration's LDS stores can't pass this iteration's LDS loads (DS in-order)
    }
}

// ---------------------------------------------------------------------------
// Kernel 3 (hot): fused per-edge MLP. 1 wave per ETILES 16-edge tiles.
//   pre  = qc[qid] + nh[head] + nt[tail]                 (gathers, L2-resident)
//   h1   = relu(pre + edge_emb @ Wr1)                    (WMMA GEMM1, K=64,N=64)
//   h2   = relu(h1 @ W2 + b2)                            (WMMA GEMM2, K=64,N=128)
//   out  = sigmoid(h2 . Wh + bh)                         (fused VALU reduce)
// ---------------------------------------------------------------------------
__global__ void __launch_bounds__(32)
edge_kernel(const float* __restrict__ edge_emb, const int* __restrict__ edge_index,
            const int* __restrict__ node_batch,
            const float* __restrict__ qc, const float* __restrict__ nh, const float* __restrict__ nt,
            const unsigned short* __restrict__ Wr1s, const unsigned short* __restrict__ W2s,
            const float* __restrict__ b2, const float* __restrict__ Wh,
            const float* __restrict__ bhp,
            float* __restrict__ out, int nE) {
    __shared__ float          accT[16 * ACC_STRIDE];
    __shared__ unsigned short h1T[16 * H1_STRIDE];
    __shared__ float          outT[16];

    const int L = threadIdx.x, g = L >> 4, n = L & 15;
    const int row2 = L >> 1, cb = (L & 1) * 32;

    // hoisted weight fragments + per-lane bias / head-weight values
    v16bf wr1f[4][2];
#pragma unroll
    for (int t = 0; t < 4; ++t) {
        wr1f[t][0] = load_b_swz(Wr1s, t, 0, L);
        wr1f[t][1] = load_b_swz(Wr1s, t, 1, L);
    }
    v16bf w2f[8][2];
    float b2v[8], whv[8];
#pragma unroll
    for (int t = 0; t < 8; ++t) {
        w2f[t][0] = load_b_swz(W2s, t, 0, L);
        w2f[t][1] = load_b_swz(W2s, t, 1, L);
        b2v[t] = b2[t * 16 + n];
        whv[t] = Wh[t * 16 + n];
    }
    const float b0 = bhp[0];

    for (int tt = 0; tt < ETILES; ++tt) {
        const int base = (blockIdx.x * ETILES + tt) * 16;
        if (base >= nE) break;  // uniform

        // ---- gather pre = qc[qid] + nh[head] + nt[tail] into LDS (f32 16x64) ----
        {
            int e = base + row2; if (e >= nE) e = nE - 1;
            int hd  = edge_index[e];
            int tl  = edge_index[nE + e];
            int qid = node_batch[hd];
            const float4* qp = (const float4*)(qc + qid * 64 + cb);
            const float4* hp = (const float4*)(nh + hd  * 64 + cb);
            const float4* tp = (const float4*)(nt + tl  * 64 + cb);
#pragma unroll
            for (int jj = 0; jj < 8; ++jj) {
                float4 a = qp[jj], b = hp[jj], c = tp[jj];
                float4 s;
                s.x = a.x + b.x + c.x; s.y = a.y + b.y + c.y;
                s.z = a.z + b.z + c.z; s.w = a.w + b.w + c.w;
                *(float4*)&accT[row2 * ACC_STRIDE + cb + jj * 4] = s;
            }
        }

        // ---- A fragments from streaming edge_embeddings read ----
        int er = base + n; if (er >= nE) er = nE - 1;
        const float* rowp = edge_emb + er * 64;
        v16bf a0 = load_a_global(rowp, 0, g);
        v16bf a1 = load_a_global(rowp, 1, g);

        // ---- GEMM1: edge_emb(16x64) @ Wr1(64x64) ----
        v8f d1[4];
#pragma unroll
        for (int t = 0; t < 4; ++t) {
            v8f d = {};
            d = wmma_bf16(a0, wr1f[t][0], d);
            d = wmma_bf16(a1, wr1f[t][1], d);
            d1[t] = d;
        }

        lds_fence();  // accT stores from all lanes complete (single wave)

        // ---- h1 = relu(pre + GEMM1) -> bf16 LDS tile ----
#pragma unroll
        for (int t = 0; t < 4; ++t)
#pragma unroll
            for (int r = 0; r < 8; ++r) {
                int m = g * 8 + r;
                float v = accT[m * ACC_STRIDE + t * 16 + n] + d1[t][r];
                v = fmaxf(v, 0.0f);
                h1T[m * H1_STRIDE + t * 16 + n] = f2bfbits(v);
            }

        lds_fence();  // h1T visible to all lanes

        // ---- GEMM2 A fragments ----
        v16bf a20 = load_a_lds(h1T, n, 0, g);
        v16bf a21 = load_a_lds(h1T, n, 1, g);

        // ---- GEMM2 (h1 @ W2) + fused bias/relu/Wh dot, one N-tile at a time ----
        float s[8] = {0.f, 0.f, 0.f, 0.f, 0.f, 0.f, 0.f, 0.f};
#pragma unroll
        for (int t = 0; t < 8; ++t) {
            v8f d = {};
            d = wmma_bf16(a20, w2f[t][0], d);
            d = wmma_bf16(a21, w2f[t][1], d);
#pragma unroll
            for (int r = 0; r < 8; ++r) {
                float h = fmaxf(d[r] + b2v[t], 0.0f);
                s[r] += h * whv[t];
            }
        }

        // ---- reduce over the 16 lanes of each half (sum over N) ----
#pragma unroll
        for (int mask = 1; mask < 16; mask <<= 1)
#pragma unroll
            for (int r = 0; r < 8; ++r) s[r] += __shfl_xor(s[r], mask, 32);

        if (n == 0) {
#pragma unroll
            for (int r = 0; r < 8; ++r) outT[g * 8 + r] = s[r];
        }
        lds_fence();

        // coalesced sigmoid + store by 16 lanes
        if (L < 16) {
            int eo = base + L;
            if (eo < nE) {
                float x = outT[L] + b0;
                out[eo] = 1.0f / (1.0f + __expf(-x));
            }
        }
        // next iteration's LDS stores can't pass this iteration's loads (DS in-order)
    }
}

// ---------------------------------------------------------------------------
// Launch
// ---------------------------------------------------------------------------
extern "C" void kernel_launch(void* const* d_in, const int* in_sizes, int n_in,
                              void* d_out, int out_size, void* d_ws, size_t ws_size,
                              hipStream_t stream) {
    const float* Q         = (const float*)d_in[0];
    const float* node_emb  = (const float*)d_in[1];
    const float* edge_emb  = (const float*)d_in[2];
    const int*   edge_idx  = (const int*)  d_in[3];
    const int*   node_bat  = (const int*)  d_in[4];
    const float* Wq = (const float*)d_in[5];  const float* bq = (const float*)d_in[6];
    const float* We = (const float*)d_in[7];  const float* be = (const float*)d_in[8];
    const float* Wr = (const float*)d_in[9];  const float* br = (const float*)d_in[10];
    const float* W1 = (const float*)d_in[11]; const float* b1 = (const float*)d_in[12];
    const float* W2 = (const float*)d_in[13]; const float* b2 = (const float*)d_in[14];
    const float* Wh = (const float*)d_in[15]; const float* bh = (const float*)d_in[16];
    float* out = (float*)d_out;

    const int nNodes = in_sizes[1] / 64;   // 50000
    const int nE     = in_sizes[2] / 64;   // 800000

    // workspace carve-up (256B aligned)
    char*  ws  = (char*)d_ws;
    size_t off = 0;
    auto carve = [&](size_t bytes) -> void* {
        void* p = ws + off;
        off += (bytes + 255) & ~(size_t)255;
        return p;
    };
    float*          qc   = (float*)carve(64 * 64 * sizeof(float));
    float*          bvh  = (float*)carve(64 * sizeof(float));
    float*          bvt  = (float*)carve(64 * sizeof(float));
    unsigned short* Wr1s = (unsigned short*)carve(64 * 64 * 2);
    unsigned short* Wehs = (unsigned short*)carve(64 * 64 * 2);
    unsigned short* Wets = (unsigned short*)carve(64 * 64 * 2);
    unsigned short* W2s  = (unsigned short*)carve(64 * 128 * 2);
    float*          nh   = (float*)carve((size_t)nNodes * 64 * sizeof(float));
    float*          nt   = (float*)carve((size_t)nNodes * 64 * sizeof(float));
    (void)ws_size; (void)n_in; (void)out_size;

    prep_kernel<<<1, 64, 0, stream>>>(Q, Wq, bq, We, be, Wr, br, W1, b1, W2,
                                      qc, bvh, bvt, Wr1s, Wehs, Wets, W2s);

    const int nTilesN = (nNodes + 15) / 16;
    node_kernel<<<(nTilesN + NTILES - 1) / NTILES, 32, 0, stream>>>(
        node_emb, Wehs, Wets, bvh, bvt, nh, nt, nNodes);

    const int nTilesE = (nE + 15) / 16;
    edge_kernel<<<(nTilesE + ETILES - 1) / ETILES, 32, 0, stream>>>(
        edge_emb, edge_idx, node_bat, qc, nh, nt, Wr1s, W2s,
        b2, Wh, bh, out, nE);
}
